// MSAModule_1726576853222
// MI455X (gfx1250) — matrix operand order repaired and checked by reading, not compiled, not run on hardware
//
#include <hip/hip_runtime.h>
#include <hip/hip_bf16.h>

// ---------------------------------------------------------------------------
// Types for CDNA5 WMMA (wave32): v_wmma_f32_16x16x32_bf16
// ---------------------------------------------------------------------------
typedef __attribute__((ext_vector_type(16))) __bf16 v16bf;
typedef __attribute__((ext_vector_type(8)))  float  v8f;

union FragBF { v16bf v; unsigned short u[16]; };

__device__ __forceinline__ unsigned short f2bf(float f) {
    unsigned u = __float_as_uint(f);
    unsigned r = u + 0x7FFFu + ((u >> 16) & 1u);   // round-to-nearest-even
    return (unsigned short)(r >> 16);
}
__device__ __forceinline__ float sigm(float x) { return 1.f / (1.f + expf(-x)); }
__device__ __forceinline__ float silu(float x) { return x * sigm(x); }

// ---------------------------------------------------------------------------
// Generic strided, 2D-batched GEMM with bf16 WMMA and fused epilogues.
//   C[b1,b2][m,n] (+)= epi( sum_k A[b1,b2][m,k] * B[b1,b2][k,n], ... )
// EPI: 0=store (opt *rowScale), 1=add (opt *rowScale), 2=sigmoid-store,
//      3=silu(acc1)*acc2 store (DUAL), 4=acc1*sigmoid(acc2) (opt *rowScale, DUAL),
//      5=OPM scatter store.
// swapDim!=0 remaps output row m -> (m%swapDim)*swapDim + m/swapDim.
// A is staged in LDS as [m][k]; B is staged TRANSPOSED as [n][k] so that both
// A- and B-fragment gathers are contiguous bf16 pairs (ds_load_2addr_b32).
// ---------------------------------------------------------------------------
struct GemmP {
    const float* A; long long sAr, sAc, sAb1, sAb2;
    const float* B; long long sBr, sBc, sBb1, sBb2;
    const float* B2;
    float* C;       long long sCr, sCc, sCb1, sCb2;
    const float* rowScale;
    int M, N, K, b1, b2, swapDim;
};

#define TM 64
#define TN 64
#define TK 32

template <int EPI, bool DUAL>
__global__ void __launch_bounds__(128) gemm_wmma(GemmP p) {
    __shared__ unsigned short As[TM][TK + 2];
    __shared__ unsigned short Bs[TN][TK + 2];    // transposed: [n][k]
    __shared__ unsigned short Bs2[TN][TK + 2];

    const int bz = blockIdx.z;
    const int b1 = bz / p.b2, b2 = bz % p.b2;
    const float* Ab  = p.A + (long long)b1 * p.sAb1 + (long long)b2 * p.sAb2;
    const float* Bb  = p.B + (long long)b1 * p.sBb1 + (long long)b2 * p.sBb2;
    const float* B2b = DUAL ? (p.B2 + (long long)b1 * p.sBb1 + (long long)b2 * p.sBb2) : nullptr;
    float* Cb = p.C + (long long)b1 * p.sCb1 + (long long)b2 * p.sCb2;

    const int m0 = blockIdx.y * TM;
    const int n0 = blockIdx.x * TN;
    const int tid = threadIdx.x;
    const int lane = tid & 31, wave = tid >> 5;
    const int wr = (wave >> 1) * 32, wc = (wave & 1) * 32;
    const int half = lane >> 4, l15 = lane & 15;

    // Uniform (scalar) fast-path check: entire tile in bounds, K multiple of TK.
    const bool full = (m0 + TM <= p.M) && (n0 + TN <= p.N) && ((p.K & (TK - 1)) == 0);

    v8f acc[2][2], acc2[2][2];
    for (int i = 0; i < 2; i++)
        for (int j = 0; j < 2; j++)
            for (int t = 0; t < 8; t++) { acc[i][j][t] = 0.f; acc2[i][j][t] = 0.f; }

    for (int k0 = 0; k0 < p.K; k0 += TK) {
        if (full) {
            // Branchless, unconditional loads (hot path).
            for (int t = tid; t < TM * TK; t += 128) {
                int r = t >> 5, c = t & 31;
                As[r][c] = f2bf(Ab[(long long)(m0 + r) * p.sAr + (long long)(k0 + c) * p.sAc]);
            }
            for (int t = tid; t < TK * TN; t += 128) {
                int r = t >> 6, c = t & 63;   // r = k index, c = n index
                long long off = (long long)(k0 + r) * p.sBr + (long long)(n0 + c) * p.sBc;
                Bs[c][r] = f2bf(Bb[off]);
                if (DUAL) Bs2[c][r] = f2bf(B2b[off]);
            }
        } else {
            // Edge path: clamp addresses (loads always issued), select-zero OOB.
            for (int t = tid; t < TM * TK; t += 128) {
                int r = t >> 5, c = t & 31;
                int gm = m0 + r, gk = k0 + c;
                int gmc = gm < p.M ? gm : p.M - 1;
                int gkc = gk < p.K ? gk : p.K - 1;
                float v = Ab[(long long)gmc * p.sAr + (long long)gkc * p.sAc];
                As[r][c] = f2bf((gm < p.M && gk < p.K) ? v : 0.f);
            }
            for (int t = tid; t < TK * TN; t += 128) {
                int r = t >> 6, c = t & 63;
                int gk = k0 + r, gn = n0 + c;
                int gkc = gk < p.K ? gk : p.K - 1;
                int gnc = gn < p.N ? gn : p.N - 1;
                bool ok = (gk < p.K && gn < p.N);
                long long off = (long long)gkc * p.sBr + (long long)gnc * p.sBc;
                float v = Bb[off];
                Bs[c][r] = f2bf(ok ? v : 0.f);
                if (DUAL) {
                    float v2 = B2b[off];
                    Bs2[c][r] = f2bf(ok ? v2 : 0.f);
                }
            }
        }
        __syncthreads();

        FragBF a[2], b[2], b2f[2];
        for (int i = 0; i < 2; i++) {
            int mr = wr + i * 16 + l15;
            for (int v = 0; v < 8; v++) {
                int kk = (v < 4) ? (2 * v + 8 * half) : (16 + 2 * (v - 4) + 8 * half);
                a[i].u[2 * v]     = As[mr][kk];
                a[i].u[2 * v + 1] = As[mr][kk + 1];
            }
        }
        for (int j = 0; j < 2; j++) {
            int nc = wc + j * 16 + l15;
            for (int v = 0; v < 8; v++) {
                int kk = (v < 4) ? (2 * v + 8 * half) : (16 + 2 * (v - 4) + 8 * half);
                b[j].u[2 * v]     = Bs[nc][kk];      // contiguous bf16 pair
                b[j].u[2 * v + 1] = Bs[nc][kk + 1];
                if (DUAL) {
                    b2f[j].u[2 * v]     = Bs2[nc][kk];
                    b2f[j].u[2 * v + 1] = Bs2[nc][kk + 1];
                }
            }
        }
        for (int i = 0; i < 2; i++)
            for (int j = 0; j < 2; j++) {
                acc[i][j] = __builtin_amdgcn_wmma_f32_16x16x32_bf16(
                    false, a[i].v, false, b[j].v, (short)0, acc[i][j], false, false);
                if (DUAL)
                    acc2[i][j] = __builtin_amdgcn_wmma_f32_16x16x32_bf16(
                        false, a[i].v, false, b2f[j].v, (short)0, acc2[i][j], false, false);
            }
        __syncthreads();
    }

    // Epilogue. C 16x16 f32 layout: VGPR r -> M = r + 8*half, N = lane&15.
    for (int i = 0; i < 2; i++)
        for (int j = 0; j < 2; j++)
            for (int r = 0; r < 8; r++) {
                int m = m0 + wr + i * 16 + r + 8 * half;
                int n = n0 + wc + j * 16 + l15;
                if (!full && (m >= p.M || n >= p.N)) continue;
                float v = acc[i][j][r];
                long long off;
                if (EPI == 5) {
                    off = (long long)(m >> 5) * 262144 + (long long)(n >> 5) * 1024
                        + (long long)(m & 31) * 32 + (n & 31);
                } else {
                    int mo = m;
                    if (p.swapDim) mo = (m % p.swapDim) * p.swapDim + (m / p.swapDim);
                    off = (long long)mo * p.sCr + (long long)n * p.sCc;
                }
                if (EPI == 0) {
                    if (p.rowScale) v *= p.rowScale[m];
                    Cb[off] = v;
                } else if (EPI == 1) {
                    if (p.rowScale) v *= p.rowScale[m];
                    Cb[off] += v;
                } else if (EPI == 2) {
                    Cb[off] = sigm(v);
                } else if (EPI == 3) {
                    Cb[off] = silu(v) * acc2[i][j][r];
                } else if (EPI == 4) {
                    float o = v * sigm(acc2[i][j][r]);
                    if (p.rowScale) o *= p.rowScale[m];
                    Cb[off] = o;
                } else if (EPI == 5) {
                    Cb[off] = v;
                }
            }
}

// ---------------------------------------------------------------------------
// LayerNorm: one block per row, blockDim == D (64 or 128).
// swapDim!=0 remaps the SOURCE row (for transposed tri-att 'tae').
// ---------------------------------------------------------------------------
__global__ void ln_kernel(const float* __restrict__ x, const float* __restrict__ g,
                          const float* __restrict__ b, float* __restrict__ o,
                          int D, int swapDim) {
    int r = blockIdx.x;
    int sr = swapDim ? ((r % swapDim) * swapDim + (r / swapDim)) : r;
    const float* xr = x + (long long)sr * D;
    int t = threadIdx.x;
    float v = xr[t];
    __shared__ float sd[128];
    sd[t] = v; __syncthreads();
    for (int s = D >> 1; s > 0; s >>= 1) { if (t < s) sd[t] += sd[t + s]; __syncthreads(); }
    float mu = sd[0] / D; __syncthreads();
    float d = v - mu;
    sd[t] = d * d; __syncthreads();
    for (int s = D >> 1; s > 0; s >>= 1) { if (t < s) sd[t] += sd[t + s]; __syncthreads(); }
    float ro = rsqrtf(sd[0] / D + 1e-5f);
    o[(long long)r * D + t] = d * ro * g[t] + b[t];
}

// pwa softmax: block = (h*256 + i), thread j over 256 keys.
__global__ void pwa_softmax_kernel(const float* __restrict__ bt,
                                   const float* __restrict__ pm,
                                   float* __restrict__ w) {
    int h = blockIdx.x >> 8;
    int i = blockIdx.x & 255;
    int j = threadIdx.x;
    float v = bt[(long long)((i << 8) + j) * 8 + h] + (1.f - pm[(i << 8) + j]) * -1e9f;
    __shared__ float sd[256];
    sd[j] = v; __syncthreads();
    for (int s = 128; s > 0; s >>= 1) { if (j < s) sd[j] = fmaxf(sd[j], sd[j + s]); __syncthreads(); }
    float mx = sd[0]; __syncthreads();
    float e = expf(v - mx);
    sd[j] = e; __syncthreads();
    for (int s = 128; s > 0; s >>= 1) { if (j < s) sd[j] += sd[j + s]; __syncthreads(); }
    w[((long long)blockIdx.x << 8) + j] = e / sd[0];
}

// tri-att softmax in-place over a 32-row-i chunk. block = ((il*4+h)*256 + q).
__global__ void tri_softmax_kernel(float* __restrict__ sc, const float* __restrict__ tb,
                                   const float* __restrict__ pm, int i0) {
    int q = blockIdx.x & 255;
    int ih = blockIdx.x >> 8;
    int h = ih & 3, il = ih >> 2;
    int ig = i0 + il;
    int k = threadIdx.x;
    long long idx = ((long long)blockIdx.x << 8) + k;
    float v = sc[idx] * 0.17677669529663687f             // 1/sqrt(32)
            + tb[(long long)((q << 8) + k) * 4 + h]
            + (pm[(ig << 8) + k] - 1.f) * 1e9f;
    __shared__ float sd[256];
    sd[k] = v; __syncthreads();
    for (int s = 128; s > 0; s >>= 1) { if (k < s) sd[k] = fmaxf(sd[k], sd[k + s]); __syncthreads(); }
    float mx = sd[0]; __syncthreads();
    float e = expf(v - mx);
    sd[k] = e; __syncthreads();
    for (int s = 128; s > 0; s >>= 1) { if (k < s) sd[k] += sd[k + s]; __syncthreads(); }
    sc[idx] = e / sd[0];
}

// Elementwise helpers.
__global__ void ew_mul_kernel(float* o, const float* a, const float* b, long long n) {
    for (long long i = blockIdx.x * 256ll + threadIdx.x; i < n; i += (long long)gridDim.x * 256)
        o[i] = a[i] * b[i];
}
__global__ void ew_addmul_kernel(float* z, const float* a, const float* b, long long n) {
    for (long long i = blockIdx.x * 256ll + threadIdx.x; i < n; i += (long long)gridDim.x * 256)
        z[i] += a[i] * b[i];
}
__global__ void ew_add_bcast_kernel(float* m, const float* s, long long n) {
    for (long long i = blockIdx.x * 256ll + threadIdx.x; i < n; i += (long long)gridDim.x * 256) {
        int c = (int)(i & 63);
        int row = (int)((i >> 6) & 255);
        m[i] += s[row * 64 + c];
    }
}
__global__ void concat_kernel(const float* msa, const float* hd, const float* dv,
                              float* o, long long n) {
    for (long long i = blockIdx.x * 256ll + threadIdx.x; i < n; i += (long long)gridDim.x * 256) {
        long long r = i / 35;
        int k = (int)(i % 35);
        o[i] = (k < 33) ? msa[r * 33 + k] : (k == 33 ? hd[r] : dv[r]);
    }
}
__global__ void pairmask_kernel(const float* tpm, float* pm) {
    int idx = blockIdx.x * 256 + threadIdx.x;
    pm[idx] = tpm[idx >> 8] * tpm[idx & 255];
}
__global__ void invnum_kernel(const float* mm, float* inv) {
    int idx = blockIdx.x * 256 + threadIdx.x;
    int i = idx >> 8, j = idx & 255;
    float s = 0.f;
    for (int t = 0; t < 128; t++) s += mm[t * 256 + i] * mm[t * 256 + j];
    inv[idx] = 1.f / fmaxf(s, 1.f);
}

// ---------------------------------------------------------------------------
// Host side
// ---------------------------------------------------------------------------
static GemmP mkg(const float* A, const float* B, float* C, int M, int N, int K) {
    GemmP p{};
    p.A = A; p.B = B; p.B2 = nullptr; p.C = C; p.rowScale = nullptr;
    p.M = M; p.N = N; p.K = K; p.b1 = 1; p.b2 = 1; p.swapDim = 0;
    p.sAr = K; p.sAc = 1; p.sAb1 = 0; p.sAb2 = 0;
    p.sBr = N; p.sBc = 1; p.sBb1 = 0; p.sBb2 = 0;
    p.sCr = N; p.sCc = 1; p.sCb1 = 0; p.sCb2 = 0;
    return p;
}

static void launch_gemm(hipStream_t st, int epi, const GemmP& p) {
    dim3 g((p.N + TN - 1) / TN, (p.M + TM - 1) / TM, (unsigned)(p.b1 * p.b2));
    dim3 b(128);
    switch (epi) {
        case 0: gemm_wmma<0, false><<<g, b, 0, st>>>(p); break;
        case 1: gemm_wmma<1, false><<<g, b, 0, st>>>(p); break;
        case 2: gemm_wmma<2, false><<<g, b, 0, st>>>(p); break;
        case 3: gemm_wmma<3, true ><<<g, b, 0, st>>>(p); break;
        case 4: gemm_wmma<4, true ><<<g, b, 0, st>>>(p); break;
        case 5: gemm_wmma<5, false><<<g, b, 0, st>>>(p); break;
    }
}

// Per-layer element counts for each of the 55 params, in dict order.
static const int PER[55] = {
    64, 64, 128, 128, 64 * 256, 64 * 256, 128 * 8, 256 * 64,        // pwa 0..7
    64, 64, 64 * 256, 64 * 256, 256 * 64,                            // mt  8..12
    64, 64, 64 * 32, 64 * 32, 1024 * 128,                            // opm 13..17
    128, 128, 128, 128, 128 * 256, 128 * 256, 128 * 128, 128 * 128,  // tmo 18..25
    128, 128, 128, 128, 128 * 256, 128 * 256, 128 * 128, 128 * 128,  // tmi 26..33
    128, 128, 128 * 4, 128 * 128, 128 * 128, 128 * 128, 128 * 128, 128 * 128, // tas 34..41
    128, 128, 128 * 4, 128 * 128, 128 * 128, 128 * 128, 128 * 128, 128 * 128, // tae 42..49
    128, 128, 128 * 512, 128 * 512, 512 * 128                        // zt  50..54
};

extern "C" void kernel_launch(void* const* d_in, const int* in_sizes, int n_in,
                              void* d_out, int out_size, void* d_ws, size_t ws_size,
                              hipStream_t stream) {
    const int S = 128, Nn = 256, Dm = 64, Dz = 128, L = 4;
    const long long MR = (long long)S * Nn;        // 32768 msa rows
    const long long ZR = (long long)Nn * Nn;       // 65536 pair rows

    const float* z0        = (const float*)d_in[0];
    const float* emb       = (const float*)d_in[1];
    const float* msa       = (const float*)d_in[2];
    const float* hd        = (const float*)d_in[3];
    const float* dv        = (const float*)d_in[4];
    const float* mmask     = (const float*)d_in[5];
    const float* tpm       = (const float*)d_in[6];
    const float* s_proj_w  = (const float*)d_in[7];
    const float* msa_proj_w= (const float*)d_in[8];
    const float* P[64];
    for (int i = 0; i < 64; i++) P[i] = nullptr;
    for (int i = 0; i + 9 < n_in && i < 55; i++) P[i] = (const float*)d_in[9 + i];

    float* ws = (float*)d_ws;
    float* z  = (float*)d_out;                                    // z lives in d_out
    float* m      = ws;                         // 32768*64
    float* mn     = ws + 2097152;               // 32768*64
    float* zn     = ws + 4194304;               // 65536*128
    float* T1     = ws + 12582912;              // 8.39M each
    float* T2     = ws + 20971520;
    float* T3     = ws + 29360128;
    float* T4     = ws + 37748736;
    float* T5     = ws + 46137344;
    float* BIG    = ws + 54525952;              // 33.55M
    float* OA     = ws + 88080384;              // 32768*32
    float* OB     = ws + 89128960;              // 32768*32
    float* BT     = ws + 90177536;              // 65536*8
    float* WSM    = ws + 90701824;              // 8*65536
    float* TB     = ws + 91226112;              // 65536*4
    float* INV    = ws + 91488256;              // 65536
    float* PM     = ws + 91553792;              // 65536
    float* SPROJ  = ws + 91619328;              // 256*64
    float* CAT    = ws + 91635712;              // 32768*35

    // ---- setup ----
    hipMemcpyAsync(z, z0, (size_t)ZR * Dz * sizeof(float), hipMemcpyDeviceToDevice, stream);
    pairmask_kernel<<<256, 256, 0, stream>>>(tpm, PM);
    invnum_kernel<<<256, 256, 0, stream>>>(mmask, INV);
    concat_kernel<<<1024, 256, 0, stream>>>(msa, hd, dv, CAT, MR * 35);
    launch_gemm(stream, 0, mkg(CAT, msa_proj_w, m, (int)MR, Dm, 35));
    launch_gemm(stream, 0, mkg(emb, s_proj_w, SPROJ, Nn, Dm, 384));
    ew_add_bcast_kernel<<<1024, 256, 0, stream>>>(m, SPROJ, MR * Dm);

    for (int l = 0; l < L; l++) {
        auto W = [&](int i) { return P[i] + (size_t)l * PER[i]; };

        // ================= pair weighted averaging =================
        ln_kernel<<<(int)MR, Dm, 0, stream>>>(m, W(0), W(1), mn, Dm, 0);
        ln_kernel<<<(int)ZR, Dz, 0, stream>>>(z, W(2), W(3), zn, Dz, 0);
        launch_gemm(stream, 0, mkg(mn, W(4), T1, (int)MR, 256, Dm));   // v
        launch_gemm(stream, 2, mkg(mn, W(5), T2, (int)MR, 256, Dm));   // g=sigmoid
        launch_gemm(stream, 0, mkg(zn, W(6), BT, (int)ZR, 8, Dz));     // b
        pwa_softmax_kernel<<<8 * 256, 256, 0, stream>>>(BT, PM, WSM);
        {   // o[h,s,i,d] = sum_j w[h,i,j] v[s,j,h,d]  -> stored as (s,i,h*32+d)
            GemmP p{};
            p.A = WSM; p.sAr = 256; p.sAc = 1; p.sAb1 = 65536; p.sAb2 = 0;
            p.B = T1;  p.sBr = 256; p.sBc = 1; p.sBb1 = 32;    p.sBb2 = 65536;
            p.B2 = nullptr; p.rowScale = nullptr;
            p.C = T3;  p.sCr = 256; p.sCc = 1; p.sCb1 = 32;    p.sCb2 = 65536;
            p.M = 256; p.N = 32; p.K = 256; p.b1 = 8; p.b2 = 128; p.swapDim = 0;
            launch_gemm(stream, 0, p);
        }
        ew_mul_kernel<<<1024, 256, 0, stream>>>(T3, T3, T2, MR * 256);
        launch_gemm(stream, 1, mkg(T3, W(7), m, (int)MR, Dm, 256));    // m += o@wo

        // ================= msa transition =================
        ln_kernel<<<(int)MR, Dm, 0, stream>>>(m, W(8), W(9), mn, Dm, 0);
        {
            GemmP p = mkg(mn, W(10), BIG, (int)MR, 256, Dm);
            p.B2 = W(11);
            launch_gemm(stream, 3, p);                                  // silu(h1)*h2
        }
        launch_gemm(stream, 1, mkg(BIG, W(12), m, (int)MR, Dm, 256));   // m += h@fc3

        // ================= outer product mean =================
        ln_kernel<<<(int)MR, Dm, 0, stream>>>(m, W(13), W(14), mn, Dm, 0);
        {
            GemmP p = mkg(mn, W(15), OA, (int)MR, 32, Dm); p.rowScale = mmask;
            launch_gemm(stream, 0, p);
            p = mkg(mn, W(16), OB, (int)MR, 32, Dm); p.rowScale = mmask;
            launch_gemm(stream, 0, p);
        }
        for (int i0 = 0; i0 < Nn; i0 += 32) {
            GemmP p{};   // zz[(i_loc,c),(j,d)] = sum_s a[s,i,c] b[s,j,d], scatter store
            p.A = OA + i0 * 32; p.sAr = 1;    p.sAc = 8192;
            p.B = OB;           p.sBr = 8192; p.sBc = 1;
            p.B2 = nullptr; p.rowScale = nullptr;
            p.C = BIG; p.sCr = 0; p.sCc = 0;
            p.sAb1 = p.sAb2 = p.sBb1 = p.sBb2 = p.sCb1 = p.sCb2 = 0;
            p.M = 1024; p.N = 8192; p.K = 128; p.b1 = 1; p.b2 = 1; p.swapDim = 0;
            launch_gemm(stream, 5, p);
            GemmP q = mkg(BIG, W(17), z + (long long)i0 * 256 * Dz, 8192, Dz, 1024);
            q.rowScale = INV + i0 * 256;
            launch_gemm(stream, 1, q);                                  // z += (zz/num)@wo
        }

        // ================= triangle multiplications =================
        for (int t = 0; t < 2; t++) {
            int b = 18 + t * 8;        // tmo then tmi
            bool incoming = (t == 1);
            ln_kernel<<<(int)ZR, Dz, 0, stream>>>(z, W(b + 0), W(b + 1), zn, Dz, 0);
            {
                GemmP p = mkg(zn, W(b + 4), BIG, (int)ZR, 256, Dz);     // a|b gated
                p.B2 = W(b + 5); p.rowScale = PM;
                launch_gemm(stream, 4, p);                              // pin*sigmoid(gin)*mask
            }
            {   // einsum over k, batched over d=128
                GemmP p{};
                if (!incoming) {    // x[i,j,d] = sum_k a[i,k,d] b[j,k,d]
                    p.A = BIG;       p.sAr = 65536; p.sAc = 256;
                    p.B = BIG + 128; p.sBr = 256;   p.sBc = 65536;
                } else {            // x[i,j,d] = sum_k a[k,i,d] b[k,j,d]
                    p.A = BIG;       p.sAr = 256;   p.sAc = 65536;
                    p.B = BIG + 128; p.sBr = 65536; p.sBc = 256;
                }
                p.sAb1 = 1; p.sBb1 = 1; p.sAb2 = p.sBb2 = 0;
                p.B2 = nullptr; p.rowScale = nullptr;
                p.C = T5; p.sCr = 32768; p.sCc = 128; p.sCb1 = 1; p.sCb2 = 0;
                p.M = 256; p.N = 256; p.K = 256; p.b1 = 128; p.b2 = 1; p.swapDim = 0;
                launch_gemm(stream, 0, p);
            }
            ln_kernel<<<(int)ZR, Dz, 0, stream>>>(T5, W(b + 2), W(b + 3), T1, Dz, 0);
            launch_gemm(stream, 0, mkg(T1, W(b + 6), T3, (int)ZR, Dz, Dz));   // ln(x)@pout
            launch_gemm(stream, 2, mkg(zn, W(b + 7), T4, (int)ZR, Dz, Dz));   // sigmoid(xn@gout)
            ew_addmul_kernel<<<2048, 256, 0, stream>>>(z, T3, T4, ZR * Dz);
        }

        // ================= triangle attentions =================
        for (int t = 0; t < 2; t++) {
            int b = 34 + t * 8;        // tas then tae
            int swap = (t == 1) ? 256 : 0;
            ln_kernel<<<(int)ZR, Dz, 0, stream>>>(z, W(b + 0), W(b + 1), zn, Dz, swap);
            launch_gemm(stream, 0, mkg(zn, W(b + 2), TB, (int)ZR, 4, Dz));    // tri bias
            launch_gemm(stream, 0, mkg(zn, W(b + 3), T1, (int)ZR, Dz, Dz));   // q
            launch_gemm(stream, 0, mkg(zn, W(b + 4), T2, (int)ZR, Dz, Dz));   // k
            launch_gemm(stream, 0, mkg(zn, W(b + 5), T3, (int)ZR, Dz, Dz));   // v
            launch_gemm(stream, 2, mkg(zn, W(b + 6), T4, (int)ZR, Dz, Dz));   // g
            for (int i0 = 0; i0 < Nn; i0 += 32) {
                GemmP p{};   // scores[il,h,q,k] = q . k
                p.A = T1 + (long long)i0 * 256 * 128;
                p.sAr = 128; p.sAc = 1;   p.sAb1 = 32768; p.sAb2 = 32;
                p.B = T2 + (long long)i0 * 256 * 128;
                p.sBr = 1;   p.sBc = 128; p.sBb1 = 32768; p.sBb2 = 32;
                p.B2 = nullptr; p.rowScale = nullptr;
                p.C = BIG; p.sCr = 256; p.sCc = 1; p.sCb1 = 262144; p.sCb2 = 65536;
                p.M = 256; p.N = 256; p.K = 32; p.b1 = 32; p.b2 = 4; p.swapDim = 0;
                launch_gemm(stream, 0, p);
                tri_softmax_kernel<<<32 * 4 * 256, 256, 0, stream>>>(BIG, TB, PM, i0);
                GemmP q{};   // o[il,q,h,d] = sum_k w * v
                q.A = BIG; q.sAr = 256; q.sAc = 1; q.sAb1 = 262144; q.sAb2 = 65536;
                q.B = T3 + (long long)i0 * 256 * 128;
                q.sBr = 128; q.sBc = 1; q.sBb1 = 32768; q.sBb2 = 32;
                q.B2 = nullptr; q.rowScale = nullptr;
                q.C = T5 + (long long)i0 * 256 * 128;
                q.sCr = 128; q.sCc = 1; q.sCb1 = 32768; q.sCb2 = 32;
                q.M = 256; q.N = 32; q.K = 256; q.b1 = 32; q.b2 = 4; q.swapDim = 0;
                launch_gemm(stream, 0, q);
            }
            ew_mul_kernel<<<2048, 256, 0, stream>>>(T5, T5, T4, ZR * Dz);
            {
                GemmP p = mkg(T5, W(b + 7), z, (int)ZR, Dz, Dz);
                p.swapDim = swap;                                       // transpose-back residual
                launch_gemm(stream, 1, p);
            }
        }

        // ================= pair transition =================
        ln_kernel<<<(int)ZR, Dz, 0, stream>>>(z, W(50), W(51), zn, Dz, 0);
        {
            GemmP p = mkg(zn, W(52), BIG, (int)ZR, 512, Dz);
            p.B2 = W(53);
            launch_gemm(stream, 3, p);                                  // silu(h1)*h2
        }
        launch_gemm(stream, 1, mkg(BIG, W(54), z, (int)ZR, Dz, 512));   // z += h@fc3
    }
    (void)in_sizes; (void)out_size; (void)ws_size;
}
